// TriangleMultiplicationOutgoing_36172214566962
// MI455X (gfx1250) — compile-verified, matrix operand order checked by reading, not run on hardware
//
#include <hip/hip_runtime.h>
#include <hip/hip_bf16.h>
#include <math.h>

typedef __attribute__((ext_vector_type(16))) _Float16 v16h;
typedef __attribute__((ext_vector_type(8)))  _Float16 v8h;
typedef __attribute__((ext_vector_type(8)))  float    v8f;
typedef __attribute__((ext_vector_type(4)))  unsigned int v4u;
typedef __attribute__((ext_vector_type(8)))  int      v8i;
typedef __attribute__((ext_vector_type(4)))  int      v4i;
typedef unsigned int       u32;
typedef unsigned long long u64;

#define NN 512
#define CC 128
#define NP (NN*NN)          // 262144 positions

union V16U { v16h v; v8h h[2]; };
union V8FU { v8f v; float f[8]; };

static __device__ __forceinline__ v16h load_v16h(const _Float16* p) {
    V16U u;
    u.h[0] = *(const v8h*)p;
    u.h[1] = *(const v8h*)(p + 8);
    return u.v;
}

static __device__ __forceinline__ v8f wmma16(v16h a, v16h b, v8f c) {
    return __builtin_amdgcn_wmma_f32_16x16x32_f16(
        /*neg_a=*/false, a, /*neg_b=*/false, b,
        /*c_mod=*/(short)0, c, /*reuse_a=*/false, /*reuse_b=*/false);
}

static __device__ __forceinline__ float sigm(float x) {
    return 1.0f / (1.0f + __expf(-x));
}

#if __has_builtin(__builtin_amdgcn_tensor_load_to_lds)
#define HAVE_TDM 1
// Issue one TDM 2D tile load: tile_rows x 32 f16, tensor row stride 512 f16,
// LDS destination padded +4 dwords per 16-dword (one row) interval -> 40 f16 rows.
static __device__ __forceinline__ void tdm_load_tile(u32 lds_byte_addr,
                                                     const _Float16* gsrc)
{
    const u64 ga = (u64)(__SIZE_TYPE__)gsrc;
    v4u g0;
    g0[0] = 1u;                                         // count=1, user descriptor
    g0[1] = lds_byte_addr;                              // LDS byte address
    g0[2] = (u32)ga;                                    // global_addr[31:0]
    g0[3] = (u32)((ga >> 32) & 0x01FFFFFFu) | (2u << 30); // global_addr[56:32] | type=2
    v8i g1;
    g1[0] = (int)((1u << 16)    // data_size = 2 bytes
                | (1u << 20)    // pad_enable
                | (3u << 22)    // pad_interval: 16 dwords (one 32-f16 row)
                | (3u << 25));  // pad_amount: 4 dwords (8 f16) -> 40-f16 LDS rows
    g1[1] = (int)(512u << 16);  // tensor_dim0[15:0]=512 in bits[63:48]
    g1[2] = (int)(512u << 16);  // tensor_dim0 hi=0, tensor_dim1[15:0]=512
    g1[3] = (int)(32u << 16);   // tensor_dim1 hi=0, tile_dim0=32 (K)
    g1[4] = (int)128u;          // tile_dim1=128 rows, tile_dim2=0
    g1[5] = (int)512u;          // tensor_dim0_stride = 512 elements
    g1[6] = 0;
    g1[7] = 0;
    const v4i z4 = {0, 0, 0, 0};
#if __clang_major__ >= 23
    const v8i z8 = {0, 0, 0, 0, 0, 0, 0, 0};
    __builtin_amdgcn_tensor_load_to_lds(g0, g1, z4, z4, z8, 0);
#else
    __builtin_amdgcn_tensor_load_to_lds(g0, g1, z4, z4, 0);
#endif
}
#else
#define HAVE_TDM 0
#endif

// ---------------------------------------------------------------------------
// k0: convert 6 weight matrices (all 128x128, stored [in][out] f32) into
//     f16 transposed [out][in] so B-fragments are per-lane contiguous-K loads.
//     Slot order: 0=Wl 1=Wlg 2=Wr 3=Wrg 4=Wg 5=Wo
// ---------------------------------------------------------------------------
__global__ __launch_bounds__(256)
void k_prep_weights(const float* __restrict__ Wl, const float* __restrict__ Wlg,
                    const float* __restrict__ Wr, const float* __restrict__ Wrg,
                    const float* __restrict__ Wg, const float* __restrict__ Wo,
                    _Float16* __restrict__ wt)
{
    int idx = blockIdx.x * 256 + threadIdx.x;      // 6*128*128 = 98304 total
    int mat = idx >> 14;
    int e   = idx & 16383;
    int o = e >> 7;
    int i = e & 127;
    const float* W = (mat == 0) ? Wl : (mat == 1) ? Wlg : (mat == 2) ? Wr
                   : (mat == 3) ? Wrg : (mat == 4) ? Wg : Wo;
    wt[idx] = (_Float16)W[i * 128 + o];
}

// ---------------------------------------------------------------------------
// k1: fused LayerNorm + 5 projections (WMMA) + mask/sigmoid epilogue.
// ---------------------------------------------------------------------------
__global__ __launch_bounds__(256)
void k_ln_proj(const float* __restrict__ act,  const float* __restrict__ mask,
               const float* __restrict__ ln_g, const float* __restrict__ ln_b,
               const float* __restrict__ bl,   const float* __restrict__ blg,
               const float* __restrict__ br,   const float* __restrict__ brg,
               const float* __restrict__ bg,
               const _Float16* __restrict__ wt,
               _Float16* __restrict__ left_t, _Float16* __restrict__ right_t,
               _Float16* __restrict__ gate_t)
{
    __shared__ _Float16 xs[16][CC + 8];
    __shared__ float    mks[16];
    const int t = threadIdx.x;
    const size_t p0 = (size_t)blockIdx.x * 16;

    {
        const int r = t >> 4, l = t & 15;
        const float* ap = act + (p0 + r) * CC + l * 8;
        float4 a0 = *(const float4*)ap;
        float4 a1 = *(const float4*)(ap + 4);
        float v[8] = {a0.x, a0.y, a0.z, a0.w, a1.x, a1.y, a1.z, a1.w};
        float s = 0.f, s2 = 0.f;
#pragma unroll
        for (int j = 0; j < 8; j++) { s += v[j]; s2 += v[j] * v[j]; }
#pragma unroll
        for (int off = 8; off > 0; off >>= 1) {
            s  += __shfl_xor(s,  off, 16);
            s2 += __shfl_xor(s2, off, 16);
        }
        const float mu = s * (1.0f / CC);
        const float rs = rsqrtf(s2 * (1.0f / CC) - mu * mu + 1e-5f);
        float4 g0 = *(const float4*)(ln_g + l * 8);
        float4 g1 = *(const float4*)(ln_g + l * 8 + 4);
        float4 b0 = *(const float4*)(ln_b + l * 8);
        float4 b1 = *(const float4*)(ln_b + l * 8 + 4);
        float gg[8] = {g0.x, g0.y, g0.z, g0.w, g1.x, g1.y, g1.z, g1.w};
        float bb[8] = {b0.x, b0.y, b0.z, b0.w, b1.x, b1.y, b1.z, b1.w};
        _Float16 xh[8];
#pragma unroll
        for (int j = 0; j < 8; j++)
            xh[j] = (_Float16)((v[j] - mu) * rs * gg[j] + bb[j]);
        *(v8h*)&xs[r][l * 8] = *(v8h*)xh;
        if (t < 16) mks[t] = mask[p0 + t];
    }
    __syncthreads();

    const int w = t >> 5, lane = t & 31;
    const int nloc = lane & 15, hi = lane >> 4;
    v16h af[4];
#pragma unroll
    for (int kc = 0; kc < 4; kc++)
        af[kc] = load_v16h(&xs[nloc][kc * 32 + hi * 16]);

    const int col = w * 16 + nloc;
    v8f acc[5];
#pragma unroll
    for (int m2 = 0; m2 < 5; m2++) {
        v8f a = {0.f, 0.f, 0.f, 0.f, 0.f, 0.f, 0.f, 0.f};
#pragma unroll
        for (int kc = 0; kc < 4; kc++) {
            v16h bf = load_v16h(wt + (size_t)m2 * 16384 + (size_t)col * CC + kc * 32 + hi * 16);
            a = wmma16(af[kc], bf, a);
        }
        acc[m2] = a;
    }

    const float blv  = bl[col],  blgv = blg[col];
    const float brv  = br[col],  brgv = brg[col];
    const float bgv  = bg[col];
    V8FU uL, uLG, uR, uRG, uG;
    uL.v = acc[0]; uLG.v = acc[1]; uR.v = acc[2]; uRG.v = acc[3]; uG.v = acc[4];
    _Float16 lh[8], rh[8], gh[8];
#pragma unroll
    for (int vv = 0; vv < 8; vv++) {
        const float mk = mks[vv + hi * 8];
        lh[vv] = (_Float16)(mk * (uL.f[vv] + blv) * sigm(uLG.f[vv] + blgv));
        rh[vv] = (_Float16)(mk * (uR.f[vv] + brv) * sigm(uRG.f[vv] + brgv));
        gh[vv] = (_Float16)(sigm(uG.f[vv] + bgv));
    }
    const size_t ob = (size_t)col * NP + p0 + hi * 8;
    *(v8h*)(left_t  + ob) = *(v8h*)lh;
    *(v8h*)(right_t + ob) = *(v8h*)rh;
    *(v8h*)(gate_t  + ob) = *(v8h*)gh;
}

// ---------------------------------------------------------------------------
// k2: out[c][i][j] = sum_k L[c][i][k] * R[c][j][k]   (128 x NT-GEMM 512^3)
//     128x128 output tile per block; 8 waves x (4x2) WMMA tiles each.
//     TDM (tensor_load_to_lds) double-buffered one K-step ahead; the DMA
//     writes padded 40-f16 LDS rows -> conflict-free fragment reads.
// ---------------------------------------------------------------------------
#define K2ROW 40     // 32 f16 + 8 f16 pad (20 dwords; gcd(20,64)=4 -> 16 banks)

__global__ __launch_bounds__(256)
void k_einsum(const _Float16* __restrict__ left_t,
              const _Float16* __restrict__ right_t,
              float* __restrict__ out_t)
{
    __shared__ _Float16 Lt[2][128][K2ROW];
    __shared__ _Float16 Rt[2][128][K2ROW];
    const int t = threadIdx.x;
    const int i0 = blockIdx.x * 128, j0 = blockIdx.y * 128;
    const size_t cb = (size_t)blockIdx.z * NP;
    const int w = t >> 5, lane = t & 31, nloc = lane & 15, hi = lane >> 4;
    const int wi = w & 1;        // row half: [wi*64, wi*64+64)
    const int wj = w >> 1;       // col group: [wj*32, wj*32+32)

    const _Float16* Lbase = left_t  + cb + (size_t)i0 * NN;   // tile row start
    const _Float16* Rbase = right_t + cb + (size_t)j0 * NN;

    v8f acc[4][2];
#pragma unroll
    for (int ii = 0; ii < 4; ii++)
#pragma unroll
        for (int jj = 0; jj < 2; jj++)
            acc[ii][jj] = (v8f){0.f,0.f,0.f,0.f,0.f,0.f,0.f,0.f};

#if HAVE_TDM
    if (w == 0) {
        tdm_load_tile((u32)(__SIZE_TYPE__)&Lt[0][0][0], Lbase);
        tdm_load_tile((u32)(__SIZE_TYPE__)&Rt[0][0][0], Rbase);
    }
    for (int s = 0; s < 16; s++) {
        const int b = s & 1;
        if (w == 0) {
            if (s + 1 < 16) {
                tdm_load_tile((u32)(__SIZE_TYPE__)&Lt[b ^ 1][0][0], Lbase + (s + 1) * 32);
                tdm_load_tile((u32)(__SIZE_TYPE__)&Rt[b ^ 1][0][0], Rbase + (s + 1) * 32);
                __builtin_amdgcn_s_wait_tensorcnt(2);   // current pair landed
            } else {
                __builtin_amdgcn_s_wait_tensorcnt(0);
            }
        }
        __syncthreads();                                 // buffer b visible to all
        v16h aF[4], bF[2];
#pragma unroll
        for (int ii = 0; ii < 4; ii++)
            aF[ii] = load_v16h(&Lt[b][wi * 64 + ii * 16 + nloc][hi * 16]);
#pragma unroll
        for (int jj = 0; jj < 2; jj++)
            bF[jj] = load_v16h(&Rt[b][wj * 32 + jj * 16 + nloc][hi * 16]);
#pragma unroll
        for (int ii = 0; ii < 4; ii++)
#pragma unroll
            for (int jj = 0; jj < 2; jj++)
                acc[ii][jj] = wmma16(aF[ii], bF[jj], acc[ii][jj]);
        __syncthreads();                                 // done reading buffer b
    }
#else
    // Fallback: cooperative synchronous loads (no TDM builtin on this toolchain).
    const int lrow = t >> 1, lseg = (t & 1) * 16;
    for (int s = 0; s < 16; s++) {
        const int b = s & 1;
        *(v8h*)&Lt[b][lrow][lseg]     = *(const v8h*)(Lbase + (size_t)lrow * NN + s * 32 + lseg);
        *(v8h*)&Lt[b][lrow][lseg + 8] = *(const v8h*)(Lbase + (size_t)lrow * NN + s * 32 + lseg + 8);
        *(v8h*)&Rt[b][lrow][lseg]     = *(const v8h*)(Rbase + (size_t)lrow * NN + s * 32 + lseg);
        *(v8h*)&Rt[b][lrow][lseg + 8] = *(const v8h*)(Rbase + (size_t)lrow * NN + s * 32 + lseg + 8);
        __syncthreads();
        v16h aF[4], bF[2];
#pragma unroll
        for (int ii = 0; ii < 4; ii++)
            aF[ii] = load_v16h(&Lt[b][wi * 64 + ii * 16 + nloc][hi * 16]);
#pragma unroll
        for (int jj = 0; jj < 2; jj++)
            bF[jj] = load_v16h(&Rt[b][wj * 32 + jj * 16 + nloc][hi * 16]);
#pragma unroll
        for (int ii = 0; ii < 4; ii++)
#pragma unroll
            for (int jj = 0; jj < 2; jj++)
                acc[ii][jj] = wmma16(aF[ii], bF[jj], acc[ii][jj]);
        __syncthreads();
    }
#endif

#pragma unroll
    for (int ii = 0; ii < 4; ii++) {
#pragma unroll
        for (int jj = 0; jj < 2; jj++) {
            V8FU u; u.v = acc[ii][jj];
#pragma unroll
            for (int vv = 0; vv < 8; vv++) {
                const size_t i = (size_t)(i0 + wi * 64 + ii * 16 + vv + hi * 8);
                out_t[cb + i * NN + j0 + wj * 32 + jj * 16 + nloc] = u.f[vv];
            }
        }
    }
}

// ---------------------------------------------------------------------------
// k3: LayerNorm over channels of out_t (channel-major), x Wo (WMMA), x gate.
// ---------------------------------------------------------------------------
__global__ __launch_bounds__(256)
void k_out(const float* __restrict__ out_t, const _Float16* __restrict__ gate_t,
           const float* __restrict__ cln_g, const float* __restrict__ cln_b,
           const float* __restrict__ bo,    const _Float16* __restrict__ wt,
           float* __restrict__ dout)
{
    __shared__ float    raw[16][CC + 4];
    __shared__ _Float16 xs[16][CC + 8];
    const int t = threadIdx.x;
    const size_t p0 = (size_t)blockIdx.x * 16;

    {
        const int r = t & 15, l = t >> 4;
#pragma unroll
        for (int j = 0; j < 8; j++) {
            const int c = l * 8 + j;
            raw[r][c] = out_t[(size_t)c * NP + p0 + r];
        }
    }
    __syncthreads();

    {
        const int r = t >> 4, l = t & 15;
        float v[8]; float s = 0.f, s2 = 0.f;
#pragma unroll
        for (int j = 0; j < 8; j++) { v[j] = raw[r][l * 8 + j]; s += v[j]; s2 += v[j] * v[j]; }
#pragma unroll
        for (int off = 8; off > 0; off >>= 1) {
            s  += __shfl_xor(s,  off, 16);
            s2 += __shfl_xor(s2, off, 16);
        }
        const float mu = s * (1.0f / CC);
        const float rs = rsqrtf(s2 * (1.0f / CC) - mu * mu + 1e-5f);
        float4 g0 = *(const float4*)(cln_g + l * 8);
        float4 g1 = *(const float4*)(cln_g + l * 8 + 4);
        float4 b0 = *(const float4*)(cln_b + l * 8);
        float4 b1 = *(const float4*)(cln_b + l * 8 + 4);
        float gg[8] = {g0.x, g0.y, g0.z, g0.w, g1.x, g1.y, g1.z, g1.w};
        float bb[8] = {b0.x, b0.y, b0.z, b0.w, b1.x, b1.y, b1.z, b1.w};
        _Float16 xh[8];
#pragma unroll
        for (int j = 0; j < 8; j++)
            xh[j] = (_Float16)((v[j] - mu) * rs * gg[j] + bb[j]);
        *(v8h*)&xs[r][l * 8] = *(v8h*)xh;
    }
    __syncthreads();

    const int w = t >> 5, lane = t & 31;
    const int nloc = lane & 15, hi = lane >> 4;
    v16h af[4];
#pragma unroll
    for (int kc = 0; kc < 4; kc++)
        af[kc] = load_v16h(&xs[nloc][kc * 32 + hi * 16]);
    const int col = w * 16 + nloc;
    v8f acc = {0.f,0.f,0.f,0.f,0.f,0.f,0.f,0.f};
#pragma unroll
    for (int kc = 0; kc < 4; kc++) {
        v16h bf = load_v16h(wt + (size_t)5 * 16384 + (size_t)col * CC + kc * 32 + hi * 16);
        acc = wmma16(af[kc], bf, acc);
    }

    const float bov = bo[col];
    _Float16 gh[8];
    *(v8h*)gh = *(const v8h*)(gate_t + (size_t)col * NP + p0 + hi * 8);
    V8FU u; u.v = acc;
#pragma unroll
    for (int vv = 0; vv < 8; vv++) {
        const size_t p = p0 + hi * 8 + vv;
        dout[p * CC + col] = (u.f[vv] + bov) * (float)gh[vv];
    }
}

// ---------------------------------------------------------------------------
// Workspace layout (bytes):
//   [0, 196608)                 wt      (6 x 128x128 f16, transposed [out][in])
//   [196608, +64MB)             left_t  f16 [c][i*N+k]
//   [+64MB,  +128MB)            right_t f16 [c][j*N+k]
//   [+128MB, +192MB)            gate_t  f16 [c][p]
//   [+192MB, +320MB)            out_t   f32 [c][i*N+j]
// ---------------------------------------------------------------------------
extern "C" void kernel_launch(void* const* d_in, const int* in_sizes, int n_in,
                              void* d_out, int out_size, void* d_ws, size_t ws_size,
                              hipStream_t stream)
{
    const float* act   = (const float*)d_in[0];
    const float* mask  = (const float*)d_in[1];
    const float* ln_g  = (const float*)d_in[2];
    const float* ln_b  = (const float*)d_in[3];
    const float* Wl    = (const float*)d_in[4];
    const float* bl    = (const float*)d_in[5];
    const float* Wr    = (const float*)d_in[6];
    const float* br    = (const float*)d_in[7];
    const float* Wlg   = (const float*)d_in[8];
    const float* blg   = (const float*)d_in[9];
    const float* Wrg   = (const float*)d_in[10];
    const float* brg   = (const float*)d_in[11];
    const float* cln_g = (const float*)d_in[12];
    const float* cln_b = (const float*)d_in[13];
    const float* Wo    = (const float*)d_in[14];
    const float* bo    = (const float*)d_in[15];
    const float* Wg    = (const float*)d_in[16];
    const float* bg    = (const float*)d_in[17];

    char* ws = (char*)d_ws;
    const size_t HALF64 = 67108864ull;       // 64 MB
    _Float16* wt      = (_Float16*)(ws);
    _Float16* left_t  = (_Float16*)(ws + 196608);
    _Float16* right_t = (_Float16*)(ws + 196608 + 1 * HALF64);
    _Float16* gate_t  = (_Float16*)(ws + 196608 + 2 * HALF64);
    float*    out_t   = (float*)   (ws + 196608 + 3 * HALF64);
    float*    dout    = (float*)d_out;

    k_prep_weights<<<384, 256, 0, stream>>>(Wl, Wlg, Wr, Wrg, Wg, Wo, wt);
    k_ln_proj<<<NP / 16, 256, 0, stream>>>(act, mask, ln_g, ln_b,
                                           bl, blg, br, brg, bg,
                                           wt, left_t, right_t, gate_t);
    k_einsum<<<dim3(NN / 128, NN / 128, CC), 256, 0, stream>>>(left_t, right_t, out_t);
    k_out<<<NP / 16, 256, 0, stream>>>(out_t, gate_t, cln_g, cln_b, bo, wt, dout);
}
